// GFE_13494787244076
// MI455X (gfx1250) — matrix operand order repaired and checked by reading, not compiled, not run on hardware
//
#include <hip/hip_runtime.h>

// ---------------------------------------------------------------------------
// MI455X (gfx1250) fused cross-attention block, bf16 WMMA everywhere.
//   y   = BN(conv1x1_in(x))            -> bf16 workspace (B,192,H,W)
//   att = {window / column / row} attn -> bf16 workspace (B,96,H,W)
//   out = conv1x1_out(att) + b_out     -> f32 d_out      (B,96,H,W)
// Matmuls: v_wmma_f32_16x16x32_bf16 (wave32); all fragments loaded as 128-bit
// chunks (ds_load_b128 / global_load_b128). Conv weights are pre-swizzled to
// fragment layout and DMA'd to LDS via the Tensor Data Mover
// (tensor_load_to_lds + s_wait_tensorcnt) when the builtin is available.
// ---------------------------------------------------------------------------

typedef __bf16 v16bf __attribute__((ext_vector_type(16)));
typedef float  v8f   __attribute__((ext_vector_type(8)));
typedef unsigned short us8 __attribute__((ext_vector_type(8)));
typedef unsigned int uint4v __attribute__((ext_vector_type(4)));
typedef int int8v __attribute__((ext_vector_type(8)));
typedef int int4v __attribute__((ext_vector_type(4)));

union ABFrag { v16bf v; unsigned short u[16]; us8 h[2]; };

#define WMMA_BF16(a, b, c) \
  __builtin_amdgcn_wmma_f32_16x16x32_bf16(false, (a), false, (b), (short)0, (c), false, false)

#if defined(__has_builtin)
#if __has_builtin(__builtin_amdgcn_tensor_load_to_lds) && \
    __has_builtin(__builtin_amdgcn_s_wait_tensorcnt)
#define USE_TDM 1
#endif
#endif

#ifdef USE_TDM
// 1-D TDM copy: nelem bf16 elements, contiguous global -> contiguous LDS.
// D# group0: count=1 | lds_addr | global_addr(57b) | type=2 ("image").
// D# group1: data_size=2B, tensor_dim0=tile_dim0=nelem, tensor_dim1=1,
//            tile_dim1=0 (1-D), dim0_stride=nelem. Groups 2/3 unused (zero).
// 6-arg builtin form (clang-23 / therock-10.0 headers).
__device__ __forceinline__ void tdm_load_1d_bf16(unsigned int lds_byte,
                                                 const void* gsrc,
                                                 unsigned int nelem) {
  const unsigned long long ga = (unsigned long long)gsrc;
  uint4v g0;
  g0.x = 1u;                                        // count = 1 (user D#)
  g0.y = lds_byte;                                  // lds_addr (bytes)
  g0.z = (unsigned int)ga;                          // global_addr[31:0]
  g0.w = (unsigned int)(ga >> 32) | (2u << 30);     // global_addr[56:32] | type=2
  int8v g1 = {};
  g1[0] = 1 << 16;                                  // data_size = 2 bytes
  g1[1] = (int)((nelem & 0xFFFFu) << 16);           // tensor_dim0[15:0]  @bit48
  g1[2] = (int)((nelem >> 16) | (1u << 16));        // tensor_dim0[31:16] | tensor_dim1=1
  g1[3] = (int)((nelem & 0xFFFFu) << 16);           // tile_dim0 @bit112
  g1[5] = (int)nelem;                               // tensor_dim0_stride[31:0]
  const int4v z4 = {};
  const int8v z8 = {};
  __builtin_amdgcn_tensor_load_to_lds(g0, g1, z4, z4, z8, 0);
}
#endif

__device__ __forceinline__ unsigned short f2bf(float f) {
  unsigned int u = __float_as_uint(f);
  unsigned int rb = ((u >> 16) & 1u) + 0x7FFFu;   // round to nearest even
  return (unsigned short)((u + rb) >> 16);
}

// CDNA5 16-bit A-matrix 16x32 element->K map (ISA 7.12.2)
__device__ __forceinline__ int a_k(int lane, int j) {
  return ((lane >> 4) * 8) + ((j >> 3) * 16) + (j & 7);
}

// Load A-frag from a row-major [row][rowStride] bf16 tile: rows=M, cols=K.
__device__ __forceinline__ void load_a_frag(ABFrag& A, const unsigned short* base,
                                            int row, int rowStride, int k0, int lane) {
  const int kb = k0 + ((lane >> 4) * 8);
  A.h[0] = *(const us8*)&base[row * rowStride + kb];
  A.h[1] = *(const us8*)&base[row * rowStride + kb + 16];
}
// Load B-frag from a row-major [col][colStride] bf16 tile: rows hold K contiguous.
__device__ __forceinline__ void load_b_frag(ABFrag& B, const unsigned short* base,
                                            int col, int colStride, int k0, int lane) {
  const int kb = k0 + ((lane >> 4) * 8);
  B.h[0] = *(const us8*)&base[col * colStride + kb];
  B.h[1] = *(const us8*)&base[col * colStride + kb + 16];
}
// Load pre-swizzled weight A-frag (fragment-layout blob, 16 elems per lane).
__device__ __forceinline__ void load_w_frag(ABFrag& A, const unsigned short* w,
                                            int mt, int ks, int lane) {
  const int base = ((mt * 3 + ks) * 32 + lane) * 16;
  A.h[0] = *(const us8*)&w[base];
  A.h[1] = *(const us8*)&w[base + 8];
}

// ---------------------------------------------------------------------------
// Kernel 0: pack weights into exact A-fragment layout (bf16) and fold BN.
// pin : 12 mtiles x 3 ksteps x 32 lanes x 16 elems (w_in, 192x96)
// pout:  6 mtiles x 3 ksteps x 32 lanes x 16 elems (w_out, 96x96)
// scale/shift: y = acc*scale[ch] + shift[ch]
// ---------------------------------------------------------------------------
__global__ void pack_kernel(const float* __restrict__ w_in,
                            const float* __restrict__ w_out,
                            const float* __restrict__ b_in,
                            const float* __restrict__ gamma,
                            const float* __restrict__ beta,
                            const float* __restrict__ mean,
                            const float* __restrict__ var,
                            unsigned short* __restrict__ pin,
                            unsigned short* __restrict__ pout,
                            float* __restrict__ scale,
                            float* __restrict__ shift) {
  const int i = blockIdx.x * 256 + threadIdx.x;
  if (i < 18432) {
    const int j = i & 15, lane = (i >> 4) & 31, ks = (i >> 9) % 3, mt = i / 1536;
    pin[i] = f2bf(w_in[(mt * 16 + (lane & 15)) * 96 + ks * 32 + a_k(lane, j)]);
  }
  if (i < 9216) {
    const int j = i & 15, lane = (i >> 4) & 31, ks = (i >> 9) % 3, mt = i / 1536;
    pout[i] = f2bf(w_out[(mt * 16 + (lane & 15)) * 96 + ks * 32 + a_k(lane, j)]);
  }
  if (i < 192) {
    const float sc = gamma[i] * rsqrtf(var[i] + 1e-5f);
    scale[i] = sc;
    shift[i] = (b_in[i] - mean[i]) * sc + beta[i];
  }
}

// ---------------------------------------------------------------------------
// Kernel 1: conv1x1 (96->192) + BN -> bf16 NCHW.
// One block = one 16-token tile staged to LDS; 4 waves x 3 ch-tiles.
// Weights (36KB, fragment layout) DMA'd to LDS once per block via TDM.
// ---------------------------------------------------------------------------
__global__ __launch_bounds__(128) void conv_in_bn_kernel(
    const float* __restrict__ x, const unsigned short* __restrict__ wpack,
    const float* __restrict__ scale, const float* __restrict__ shift,
    unsigned short* __restrict__ y) {
  __shared__ __align__(16) unsigned short xs[16 * 96];   // [token][channel]
#ifdef USE_TDM
  __shared__ __align__(16) unsigned short wlds[18432];   // fragment-layout weights
#endif
  const int tid = threadIdx.x, lane = tid & 31, wave = tid >> 5;
  const int n0 = blockIdx.x * 16;
  const int b  = n0 / 25600;
  const int hw0 = n0 % 25600;

#ifdef USE_TDM
  if (wave == 0)
    tdm_load_1d_bf16((unsigned int)(unsigned long long)(void*)wlds, wpack, 18432);
#endif
  if (tid < 96 && blockIdx.x + 1 < gridDim.x) {          // prefetch next tile
    const int nn = n0 + 16;
    __builtin_prefetch(&x[(size_t)((nn / 25600) * 96 + tid) * 25600 + nn % 25600], 0, 1);
  }
  for (int i = tid; i < 96 * 16; i += 128) {             // coalesced along tokens
    const int c = i >> 4, t = i & 15;
    xs[t * 96 + c] = f2bf(x[(size_t)(b * 96 + c) * 25600 + hw0 + t]);
  }
#ifdef USE_TDM
  if (wave == 0) __builtin_amdgcn_s_wait_tensorcnt(0);
#endif
  __syncthreads();

  ABFrag Bf[3];                                          // token frags, reused 3x
#pragma unroll
  for (int ks = 0; ks < 3; ++ks)
    load_b_frag(Bf[ks], xs, lane & 15, 96, ks * 32, lane);

#pragma unroll
  for (int i = 0; i < 3; ++i) {
    const int mt = wave * 3 + i;                         // 0..11
    v8f acc = {};
#pragma unroll
    for (int ks = 0; ks < 3; ++ks) {
      ABFrag A;
#ifdef USE_TDM
      load_w_frag(A, wlds, mt, ks, lane);
#else
      load_w_frag(A, wpack, mt, ks, lane);
#endif
      acc = WMMA_BF16(A.v, Bf[ks].v, acc);
    }
    const int hw = hw0 + (lane & 15);
#pragma unroll
    for (int r = 0; r < 8; ++r) {
      const int ch = mt * 16 + r + 8 * (lane >> 4);
      y[(size_t)(b * 192 + ch) * 25600 + hw] = f2bf(acc[r] * scale[ch] + shift[ch]);
    }
  }
}

// ---------------------------------------------------------------------------
// Kernel 2: 5x5 window attention (branch 0). One wave per window.
// L=25 padded to 32, d=32. Qs row-major [t][c]; Vt transposed [c][t].
// ---------------------------------------------------------------------------
__global__ __launch_bounds__(32) void win_attn_kernel(
    const unsigned short* __restrict__ y, unsigned short* __restrict__ o) {
  __shared__ __align__(16) unsigned short Qs[32 * 32];
  __shared__ __align__(16) unsigned short Vt[32 * 32];
  __shared__ __align__(16) float          Sf[32 * 33];
  __shared__ __align__(16) unsigned short As[32 * 32];

  const int lane = threadIdx.x;
  const int wid = blockIdx.x;
  const int b  = wid >> 10;
  const int wh = (wid >> 5) & 31;
  const int ww = wid & 31;

  for (int i = lane; i < 1024; i += 32) { Qs[i] = 0; Vt[i] = 0; As[i] = 0; }
  __syncthreads();
  for (int i = lane; i < 25 * 32; i += 32) {
    const int t = i >> 5, c = i & 31;
    const int h = wh * 5 + t / 5, w = ww * 5 + t % 5;
    const size_t base = ((size_t)(b * 192 + c) * 160 + h) * 160 + w;
    Qs[t * 32 + c] = y[base];
    Vt[c * 32 + t] = y[base + (size_t)32 * 25600];
  }
  __syncthreads();

  // S = Q * Q^T (K=32, one WMMA per 16x16 tile)
#pragma unroll
  for (int ti = 0; ti < 2; ++ti)
#pragma unroll
    for (int tj = 0; tj < 2; ++tj) {
      ABFrag A, Bf;
      load_a_frag(A, Qs, ti * 16 + (lane & 15), 32, 0, lane);
      load_b_frag(Bf, Qs, tj * 16 + (lane & 15), 32, 0, lane);
      v8f c = {};
      c = WMMA_BF16(A.v, Bf.v, c);
#pragma unroll
      for (int r = 0; r < 8; ++r)
        Sf[(ti * 16 + r + 8 * (lane >> 4)) * 33 + tj * 16 + (lane & 15)] = c[r];
    }
  __syncthreads();

  if (lane < 25) {
    float m = -1e30f;
    for (int c = 0; c < 25; ++c) m = fmaxf(m, Sf[lane * 33 + c]);
    float s = 0.f;
    for (int c = 0; c < 25; ++c) s += __expf(Sf[lane * 33 + c] - m);
    const float inv = 1.f / s;
    for (int c = 0; c < 25; ++c)
      As[lane * 32 + c] = f2bf(__expf(Sf[lane * 33 + c] - m) * inv);
  }
  __syncthreads();

  // O = A * V (K=32)
#pragma unroll
  for (int ti = 0; ti < 2; ++ti)
#pragma unroll
    for (int cj = 0; cj < 2; ++cj) {
      ABFrag A, Bf;
      load_a_frag(A, As, ti * 16 + (lane & 15), 32, 0, lane);
      load_b_frag(Bf, Vt, cj * 16 + (lane & 15), 32, 0, lane);
      v8f c = {};
      c = WMMA_BF16(A.v, Bf.v, c);
#pragma unroll
      for (int r = 0; r < 8; ++r) {
        const int t = ti * 16 + r + 8 * (lane >> 4);
        if (t < 25) {
          const int h = wh * 5 + t / 5, w = ww * 5 + t % 5;
          const int ch = cj * 16 + (lane & 15);
          o[((size_t)(b * 96 + ch) * 160 + h) * 160 + w] = f2bf(c[r]);
        }
      }
    }
}

// ---------------------------------------------------------------------------
// Kernel 3: strip attention (branch 0 = columns, 1 = rows).
// One 128-thread block per (strip, 32-query band). L=160, d=32.
// Qs [t][32] row-major; Vt [c][160] transposed; As [q][160] row-major.
// ---------------------------------------------------------------------------
__global__ __launch_bounds__(128) void strip_attn_kernel(
    const unsigned short* __restrict__ y, unsigned short* __restrict__ o,
    int branch) {
  __shared__ __align__(16) unsigned short Qs[160 * 32];
  __shared__ __align__(16) unsigned short Vt[32 * 160];
  __shared__ __align__(16) float          Sf[32 * 161];
  __shared__ __align__(16) unsigned short As[32 * 160];

  const int tid = threadIdx.x;
  const int lane = tid & 31;
  const int wave = tid >> 5;
  const int strip = blockIdx.x;                  // 0..1279 = b*160 + fixed
  const int band = blockIdx.y;                   // 0..4 (32 queries each)
  const int b = strip / 160;
  const int fix = strip % 160;
  const int qbase = 64 + branch * 64;            // q channel base in y
  const int obase = 32 + branch * 32;            // output channel base

  auto pix = [&](int t) -> size_t {              // pixel offset for token t
    return branch == 0 ? (size_t)t * 160 + fix : (size_t)fix * 160 + t;
  };

  for (int i = tid; i < 160 * 32; i += 128) {    // c-major: coalesced in t
    const int c = i / 160, t = i % 160;
    const size_t base = (size_t)(b * 192 + qbase + c) * 25600 + pix(t);
    Qs[t * 32 + c] = y[base];
    Vt[c * 160 + t] = y[base + (size_t)32 * 25600];
  }
  __syncthreads();

  // S band: 2x10 tiles of 16x16, 5 tiles per wave, K=32.
  for (int i = 0; i < 5; ++i) {
    const int tile = wave * 5 + i;
    const int ti = tile / 10, tj = tile % 10;
    ABFrag A, Bf;
    load_a_frag(A, Qs, band * 32 + ti * 16 + (lane & 15), 32, 0, lane);
    load_b_frag(Bf, Qs, tj * 16 + (lane & 15), 32, 0, lane);
    v8f c = {};
    c = WMMA_BF16(A.v, Bf.v, c);
#pragma unroll
    for (int r = 0; r < 8; ++r)
      Sf[(ti * 16 + r + 8 * (lane >> 4)) * 161 + tj * 16 + (lane & 15)] = c[r];
  }
  __syncthreads();

  if (tid < 32) {
    float m = -1e30f;
    for (int c = 0; c < 160; ++c) m = fmaxf(m, Sf[tid * 161 + c]);
    float s = 0.f;
    for (int c = 0; c < 160; ++c) s += __expf(Sf[tid * 161 + c] - m);
    const float inv = 1.f / s;
    for (int c = 0; c < 160; ++c)
      As[tid * 160 + c] = f2bf(__expf(Sf[tid * 161 + c] - m) * inv);
  }
  __syncthreads();

  // O = A(32x160) x V(160x32): 4 output tiles, one per wave, K=160.
  {
    const int ti = wave >> 1, cj = wave & 1;
    v8f c = {};
#pragma unroll
    for (int k0 = 0; k0 < 160; k0 += 32) {
      ABFrag A, Bf;
      load_a_frag(A, As, ti * 16 + (lane & 15), 160, k0, lane);
      load_b_frag(Bf, Vt, cj * 16 + (lane & 15), 160, k0, lane);
      c = WMMA_BF16(A.v, Bf.v, c);
    }
#pragma unroll
    for (int r = 0; r < 8; ++r) {
      const int t = band * 32 + ti * 16 + r + 8 * (lane >> 4);
      const int ch = obase + cj * 16 + (lane & 15);
      o[(size_t)(b * 96 + ch) * 25600 + pix(t)] = f2bf(c[r]);
    }
  }
}

// ---------------------------------------------------------------------------
// Kernel 4: conv1x1 (96->96) from bf16 attention output -> f32 + bias.
// One block = one 16-token tile staged to LDS; 2 waves x 3 ch-tiles.
// Weights (18KB, fragment layout) DMA'd to LDS once per block via TDM.
// ---------------------------------------------------------------------------
__global__ __launch_bounds__(64) void conv_out_kernel(
    const unsigned short* __restrict__ a, const unsigned short* __restrict__ wpack,
    const float* __restrict__ b_out, float* __restrict__ out) {
  __shared__ __align__(16) unsigned short xs[16 * 96];   // [token][channel]
#ifdef USE_TDM
  __shared__ __align__(16) unsigned short wlds[9216];
#endif
  const int tid = threadIdx.x, lane = tid & 31, wave = tid >> 5;
  const int n0 = blockIdx.x * 16;
  const int b  = n0 / 25600;
  const int hw0 = n0 % 25600;

#ifdef USE_TDM
  if (wave == 0)
    tdm_load_1d_bf16((unsigned int)(unsigned long long)(void*)wlds, wpack, 9216);
#endif
  if (tid < 64 && blockIdx.x + 1 < gridDim.x) {
    const int nn = n0 + 16;
    __builtin_prefetch(&a[(size_t)((nn / 25600) * 96 + tid) * 25600 + nn % 25600], 0, 1);
  }
  for (int i = tid; i < 96 * 16; i += 64) {
    const int c = i >> 4, t = i & 15;
    xs[t * 96 + c] = a[(size_t)(b * 96 + c) * 25600 + hw0 + t];
  }
#ifdef USE_TDM
  if (wave == 0) __builtin_amdgcn_s_wait_tensorcnt(0);
#endif
  __syncthreads();

  ABFrag Bf[3];
#pragma unroll
  for (int ks = 0; ks < 3; ++ks)
    load_b_frag(Bf[ks], xs, lane & 15, 96, ks * 32, lane);

#pragma unroll
  for (int i = 0; i < 3; ++i) {
    const int mt = wave * 3 + i;                         // 0..5
    v8f acc = {};
#pragma unroll
    for (int ks = 0; ks < 3; ++ks) {
      ABFrag A;
#ifdef USE_TDM
      load_w_frag(A, wlds, mt, ks, lane);
#else
      load_w_frag(A, wpack, mt, ks, lane);
#endif
      acc = WMMA_BF16(A.v, Bf[ks].v, acc);
    }
    const int hw = hw0 + (lane & 15);
#pragma unroll
    for (int r = 0; r < 8; ++r) {
      const int ch = mt * 16 + r + 8 * (lane >> 4);
      out[(size_t)(b * 96 + ch) * 25600 + hw] = acc[r] + b_out[ch];
    }
  }
}

// ---------------------------------------------------------------------------
extern "C" void kernel_launch(void* const* d_in, const int* in_sizes, int n_in,
                              void* d_out, int out_size, void* d_ws, size_t ws_size,
                              hipStream_t stream) {
  const float* x     = (const float*)d_in[0];
  const float* w_in  = (const float*)d_in[1];
  const float* b_in  = (const float*)d_in[2];
  const float* gamma = (const float*)d_in[3];
  const float* beta  = (const float*)d_in[4];
  const float* mean  = (const float*)d_in[5];
  const float* var   = (const float*)d_in[6];
  const float* w_out = (const float*)d_in[7];
  const float* b_out = (const float*)d_in[8];
  float* out = (float*)d_out;

  // Workspace layout (all 16B aligned):
  //   y bf16 (8*192*160*160) | attn bf16 (8*96*160*160) |
  //   wpack_in (18432 us) | wpack_out (9216 us) | scale f32[192] | shift f32[192]
  unsigned short* y_ws    = (unsigned short*)d_ws;
  unsigned short* attn_ws = y_ws + (size_t)39321600;
  unsigned short* wpin    = attn_ws + (size_t)19660800;
  unsigned short* wpout   = wpin + 18432;
  float* scale = (float*)(wpout + 9216);
  float* shift = scale + 192;

  // 0) weight pre-swizzle + BN folding
  pack_kernel<<<72, 256, 0, stream>>>(w_in, w_out, b_in, gamma, beta, mean, var,
                                      wpin, wpout, scale, shift);

  // 1) conv_in + BN: 204800 tokens / 16 = 12800 tiles, 4 waves x 3 ch-tiles
  conv_in_bn_kernel<<<12800, 128, 0, stream>>>(x, wpin, scale, shift, y_ws);

  // 2) 5x5 window attention: 8192 windows, one wave each
  win_attn_kernel<<<8192, 32, 0, stream>>>(y_ws, attn_ws);

  // 3) column strips (branch 0) and row strips (branch 1)
  strip_attn_kernel<<<dim3(1280, 5), 128, 0, stream>>>(y_ws, attn_ws, 0);
  strip_attn_kernel<<<dim3(1280, 5), 128, 0, stream>>>(y_ws, attn_ws, 1);

  // 4) conv_out: 12800 token tiles, 2 waves x 3 ch-tiles
  conv_out_kernel<<<12800, 64, 0, stream>>>(attn_ws, wpout, b_out, out);
}
